// MLSTMBlock_35167192219983
// MI455X (gfx1250) — matrix operand order repaired and checked by reading, not compiled, not run on hardware
//
#include <hip/hip_runtime.h>
#include <hip/hip_bf16.h>

// Problem constants from the reference.
#define BB 4
#define SS 2048
#define DD 1024
#define HH 3072

typedef __attribute__((ext_vector_type(16))) __bf16 bf16x16;
typedef __attribute__((ext_vector_type(8)))  __bf16 bf16x8;
typedef __attribute__((ext_vector_type(8)))  float  f32x8;
typedef __attribute__((ext_vector_type(4)))  int    i32x4;

static __device__ __forceinline__ bf16x16 join8(bf16x8 lo, bf16x8 hi) {
  return __builtin_shufflevector(lo, hi, 0,1,2,3,4,5,6,7,8,9,10,11,12,13,14,15);
}

// ---------------------------------------------------------------------------
// CDNA5 async global->LDS staging (ASYNCcnt path), with sync fallback.
// Builtin takes i32x4* pointers: AS1 source, AS3 destination.
// ---------------------------------------------------------------------------
#if defined(__has_builtin)
#if __has_builtin(__builtin_amdgcn_global_load_async_to_lds_b128) && \
    __has_builtin(__builtin_amdgcn_s_wait_asynccnt)
#define HAS_ASYNC_LDS 1
#endif
#endif
#ifndef HAS_ASYNC_LDS
#define HAS_ASYNC_LDS 0
#endif

#if HAS_ASYNC_LDS
typedef __attribute__((address_space(1))) i32x4* gvec_ptr;
typedef __attribute__((address_space(3))) i32x4* lvec_ptr;
static __device__ __forceinline__ void async_copy16(void* lds, const void* g) {
  __builtin_amdgcn_global_load_async_to_lds_b128((gvec_ptr)g, (lvec_ptr)lds,
                                                 0, 0);
}
template <int N>
static __device__ __forceinline__ void async_wait_le() {
  __builtin_amdgcn_s_wait_asynccnt(N);
}
#else
static __device__ __forceinline__ void async_copy16(void* lds, const void* g) {
  *(bf16x8*)lds = *(const bf16x8*)g;
}
template <int N>
static __device__ __forceinline__ void async_wait_le() {}
#endif

// ---------------------------------------------------------------------------
// Fused f32 -> bf16 transpose for weights: in[K][N] -> out[N][K].
// ---------------------------------------------------------------------------
__global__ __launch_bounds__(256) void cvt_tr_kernel(
    const float* __restrict__ in, __bf16* __restrict__ out, int K, int N) {
  __shared__ float tile[32][33];
  const int tid = threadIdx.x;
  const int tx = tid & 31, ty = tid >> 5;  // 32 x 8
  const int k0 = blockIdx.y * 32, n0 = blockIdx.x * 32;
#pragma unroll
  for (int i = 0; i < 4; ++i)
    tile[ty + i * 8][tx] = in[(size_t)(k0 + ty + i * 8) * N + n0 + tx];
  __syncthreads();
#pragma unroll
  for (int i = 0; i < 4; ++i)
    out[(size_t)(n0 + ty + i * 8) * K + k0 + tx] = (__bf16)tile[tx][ty + i * 8];
}

// ---------------------------------------------------------------------------
// LayerNorm over D=1024 (one block per row), optionally fusing the two gate
// matvecs (h.Wf, h.Wi).
// ---------------------------------------------------------------------------
__global__ __launch_bounds__(256) void ln_kernel(
    const float* __restrict__ xin, const float* __restrict__ gamma,
    const float* __restrict__ beta, __bf16* __restrict__ hout,
    const float* __restrict__ Wf, const float* __restrict__ Wi,
    float* __restrict__ fl, float* __restrict__ il, int gates) {
  __shared__ float s1[256];
  __shared__ float s2[256];
  const int row = blockIdx.x;
  const int tid = threadIdx.x;
  const float* xr = xin + (size_t)row * DD;

  float xv[4];
  float sum = 0.f, sq = 0.f;
#pragma unroll
  for (int k = 0; k < 4; ++k) {
    xv[k] = xr[tid + k * 256];
    sum += xv[k];
    sq += xv[k] * xv[k];
  }
  s1[tid] = sum; s2[tid] = sq;
  __syncthreads();
  for (int off = 128; off > 0; off >>= 1) {
    if (tid < off) { s1[tid] += s1[tid + off]; s2[tid] += s2[tid + off]; }
    __syncthreads();
  }
  const float mean = s1[0] * (1.f / DD);
  const float var  = s2[0] * (1.f / DD) - mean * mean;
  const float rstd = rsqrtf(var + 1e-5f);
  __syncthreads();

  float pf = 0.f, pi = 0.f;
#pragma unroll
  for (int k = 0; k < 4; ++k) {
    int d = tid + k * 256;
    float h = (xv[k] - mean) * rstd * gamma[d] + beta[d];
    hout[(size_t)row * DD + d] = (__bf16)h;
    if (gates) { pf += h * Wf[d]; pi += h * Wi[d]; }
  }
  if (gates) {
    s1[tid] = pf; s2[tid] = pi;
    __syncthreads();
    for (int off = 128; off > 0; off >>= 1) {
      if (tid < off) { s1[tid] += s1[tid + off]; s2[tid] += s2[tid + off]; }
      __syncthreads();
    }
    if (tid == 0) { fl[row] = s1[0]; il[row] = s2[0]; }
  }
}

// ---------------------------------------------------------------------------
// Per-batch scan: lfcum = cumsum(logsigmoid(f)), w_s = logsigmoid(i) - lfcum
// ---------------------------------------------------------------------------
static __device__ __forceinline__ float logsig(float z) {
  return z >= 0.f ? -log1pf(__expf(-z)) : z - log1pf(__expf(z));
}

__global__ __launch_bounds__(256) void gate_scan_kernel(
    const float* __restrict__ fl, const float* __restrict__ il,
    const float* __restrict__ bfp, const float* __restrict__ bip,
    float* __restrict__ lfcum, float* __restrict__ wvec) {
  __shared__ float sc[256];
  const int b = blockIdx.x;
  const int tid = threadIdx.x;
  const float bf_ = bfp[0], bi_ = bip[0];
  const int base = b * SS + tid * 8;

  float lsf[8];
  float run = 0.f;
#pragma unroll
  for (int i = 0; i < 8; ++i) {
    run += logsig(fl[base + i] + bf_);
    lsf[i] = run;
  }
  sc[tid] = run;
  __syncthreads();
  for (int off = 1; off < 256; off <<= 1) {
    float v = (tid >= off) ? sc[tid - off] : 0.f;
    __syncthreads();
    sc[tid] += v;
    __syncthreads();
  }
  const float excl = (tid > 0) ? sc[tid - 1] : 0.f;
#pragma unroll
  for (int i = 0; i < 8; ++i) {
    float cum = excl + lsf[i];
    lfcum[base + i] = cum;
    wvec[base + i] = logsig(il[base + i] + bi_) - cum;
  }
}

// ---------------------------------------------------------------------------
// bf16 WMMA GEMM with double-buffered async LDS staging:
//   out = act(A[MxK] @ Wt^T + bias) (+addsrc), Wt pre-transposed [N][K].
// Block tile 128x64, 8 waves; each wave: 16 rows x 64 cols = 4 WMMA accs.
// Pipeline: issue chunk c+1's async copies into the other buffer, then
// s_wait_asynccnt <= 3 so only chunk c is guaranteed landed (DMA overlaps
// the WMMAs). OUT_MODE: 0 = f32, 1 = bf16, 2 = bf16 transposed [B][N][S].
// ---------------------------------------------------------------------------
template <int ACT, int OUT_MODE, bool ADD>
__global__ __launch_bounds__(256) void gemm_bf16_kernel(
    const __bf16* __restrict__ A, const __bf16* __restrict__ Wt,
    const float* __restrict__ bias, const float* __restrict__ addsrc,
    float* __restrict__ outF, __bf16* __restrict__ outB,
    int M, int K, int N) {
  __shared__ __align__(16) __bf16 lA[2][128][40];  // [buf][m][k]
  __shared__ __align__(16) __bf16 lB[2][64][40];   // [buf][n][k]

  const int tid  = threadIdx.x;
  const int lane = tid & 31;
  const int wvid = tid >> 5;   // 0..7 -> 16-row slice
  const int half = lane >> 4;
  const int l16  = lane & 15;

  const int m0 = blockIdx.y * 128;
  const int n0 = blockIdx.x * 64;

  const int aRow = tid >> 1, aPart = (tid & 1) * 16;  // 32B of A per thread
  const int bRow = tid >> 2, bPart = (tid & 3) * 8;   // 16B of B per thread

  auto stage = [&](int buf, int k0) {
    const __bf16* ga = A + (size_t)(m0 + aRow) * K + k0 + aPart;
    const __bf16* gb = Wt + (size_t)(n0 + bRow) * K + k0 + bPart;
    async_copy16(&lA[buf][aRow][aPart], ga);
    async_copy16(&lA[buf][aRow][aPart + 8], ga + 8);
    async_copy16(&lB[buf][bRow][bPart], gb);
  };

  f32x8 acc[4] = {};
  const int nk = K >> 5;
  stage(0, 0);

  for (int c = 0; c < nk; ++c) {
    const int cur = c & 1;
    if (c + 1 < nk) {
      stage(cur ^ 1, (c + 1) * 32);  // prefetch behind the compute
      async_wait_le<3>();            // only chunk c guaranteed complete
    } else {
      async_wait_le<0>();
    }
    __syncthreads();

    const int ar = wvid * 16 + l16;
    bf16x16 afrag = join8(*(const bf16x8*)&lA[cur][ar][half * 8],
                          *(const bf16x8*)&lA[cur][ar][16 + half * 8]);
#pragma unroll
    for (int t = 0; t < 4; ++t) {
      const int nr = t * 16 + l16;
      bf16x16 bfrag = join8(*(const bf16x8*)&lB[cur][nr][half * 8],
                            *(const bf16x8*)&lB[cur][nr][16 + half * 8]);
      acc[t] = __builtin_amdgcn_wmma_f32_16x16x32_bf16(
          false, afrag, false, bfrag, (short)0, acc[t], false, false);
    }
    __syncthreads();  // all reads of buf `cur` done before it is re-staged
  }

#pragma unroll
  for (int t = 0; t < 4; ++t) {
    const int col = n0 + t * 16 + l16;
    const float bcol = bias[col];
#pragma unroll
    for (int j = 0; j < 8; ++j) {
      const int row = m0 + wvid * 16 + half * 8 + j;  // C: VGPR j -> M row
      float v = acc[t][j] + bcol;
      if (ACT == 1) {
        v = 1.f / (1.f + __expf(-v));           // sigmoid
      } else if (ACT == 2) {
        float u = v > 0.f ? v : 0.01f * v;      // LeakyReLU then square
        v = u * u;
      }
      if (OUT_MODE == 2) {
        const int bb = row >> 11;               // row / SS
        const int s  = row & (SS - 1);
        outB[((size_t)(bb * N + col)) * SS + s] = (__bf16)v;
      } else if (OUT_MODE == 1) {
        outB[(size_t)row * N + col] = (__bf16)v;
      } else {
        size_t idx = (size_t)row * N + col;
        if (ADD) v += addsrc[idx];
        outF[idx] = v;
      }
    }
  }
}

// ---------------------------------------------------------------------------
// Fused causal mLSTM attention over transposed v (vT: [B][D][S]):
//   e[t,s] = exp(lfcum[t] + w[s]) for s<=t (exponent <= 0, stable)
//   attn[t,d] = sum_s e[t,s] vT[d][s] / (sum_s e[t,s] + 1e-6)
//   xout = x + o * attn
// Block = 16 rows x 256 cols; K-chunks of 32; double-buffered async v tiles.
// The causally-masked (diagonal) chunk is always the LAST chunk for a row
// tile, so the bulk computes exp unconditionally (uniform branch only).
// ---------------------------------------------------------------------------
__global__ __launch_bounds__(256) void attn_kernel(
    const __bf16* __restrict__ vT, const float* __restrict__ lfcum,
    const float* __restrict__ wvec, const float* __restrict__ x,
    const float* __restrict__ og, float* __restrict__ xout) {
  __shared__ __align__(16) __bf16 lE[16][40];       // score tile [t][s]
  __shared__ __align__(16) __bf16 lV[2][256][40];   // [buf][d][s]
  __shared__ float aT[16];
  __shared__ float Zs[16];

  const int tid  = threadIdx.x;
  const int lane = tid & 31;
  const int wvid = tid >> 5;  // wave -> 32-column slice
  const int half = lane >> 4;
  const int l16  = lane & 15;

  const int b  = blockIdx.z;
  const int t0 = blockIdx.y * 16;
  const int d0 = blockIdx.x * 256;

  if (tid < 16) {
    aT[tid] = lfcum[b * SS + t0 + tid];
    Zs[tid] = 0.f;
  }
  __syncthreads();

  const int et = tid >> 4;        // score row 0..15
  const int es = (tid & 15) * 2;  // score col pair (even)
  const __bf16* vrow = vT + ((size_t)(b * DD + d0 + tid)) * SS;

  auto stage = [&](int buf, int s0) {
    async_copy16(&lV[buf][tid][0],  vrow + s0);
    async_copy16(&lV[buf][tid][8],  vrow + s0 + 8);
    async_copy16(&lV[buf][tid][16], vrow + s0 + 16);
    async_copy16(&lV[buf][tid][24], vrow + s0 + 24);
  };

  f32x8 acc0 = {}, acc1 = {};
  const int nch = (t0 + 47) >> 5;  // chunks covering s in [0, t0+15]
  stage(0, 0);

  for (int c = 0; c < nch; ++c) {
    const int cur = c & 1;
    const int s0 = c * 32;
    {  // exp-score tile (bf16 pair store) + ds_add_f32 row sums
      const float at = aT[et];
      const int s = s0 + es;
      float e0, e1;
      if (c + 1 < nch) {  // uniform: interior chunk, no causal mask needed
        e0 = __expf(at + wvec[b * SS + s]);
        e1 = __expf(at + wvec[b * SS + s + 1]);
      } else {            // uniform: diagonal chunk, apply causal mask
        e0 = (s <= t0 + et)     ? __expf(at + wvec[b * SS + s])     : 0.f;
        e1 = (s + 1 <= t0 + et) ? __expf(at + wvec[b * SS + s + 1]) : 0.f;
      }
      union { __bf16 h[2]; unsigned u; } pk;
      pk.h[0] = (__bf16)e0;
      pk.h[1] = (__bf16)e1;
      *(unsigned*)&lE[et][es] = pk.u;
      atomicAdd(&Zs[et], e0 + e1);
    }
    if (c + 1 < nch) {
      stage(cur ^ 1, (c + 1) * 32);  // prefetch next v tile behind compute
      async_wait_le<4>();
    } else {
      async_wait_le<0>();
    }
    __syncthreads();

    bf16x16 afrag = join8(*(const bf16x8*)&lE[l16][half * 8],
                          *(const bf16x8*)&lE[l16][16 + half * 8]);
    const int nr = wvid * 32 + l16;
    bf16x16 bfrag0 = join8(*(const bf16x8*)&lV[cur][nr][half * 8],
                           *(const bf16x8*)&lV[cur][nr][16 + half * 8]);
    bf16x16 bfrag1 = join8(*(const bf16x8*)&lV[cur][nr + 16][half * 8],
                           *(const bf16x8*)&lV[cur][nr + 16][16 + half * 8]);
    acc0 = __builtin_amdgcn_wmma_f32_16x16x32_bf16(
        false, afrag, false, bfrag0, (short)0, acc0, false, false);
    acc1 = __builtin_amdgcn_wmma_f32_16x16x32_bf16(
        false, afrag, false, bfrag1, (short)0, acc1, false, false);
    __syncthreads();
  }

#pragma unroll
  for (int t2 = 0; t2 < 2; ++t2) {
    const int col = d0 + wvid * 32 + t2 * 16 + l16;
#pragma unroll
    for (int j = 0; j < 8; ++j) {
      const int r = half * 8 + j;
      const size_t idx = ((size_t)(b * SS + t0 + r)) * DD + col;
      const float a = (t2 ? acc1[j] : acc0[j]) / (Zs[r] + 1e-6f);
      xout[idx] = x[idx] + og[idx] * a;
    }
  }
}

// ---------------------------------------------------------------------------
// Launcher. q/k projections are dead code in the reference (output never
// depends on them) and are skipped.
// ---------------------------------------------------------------------------
extern "C" void kernel_launch(void* const* d_in, const int* in_sizes, int n_in,
                              void* d_out, int out_size, void* d_ws,
                              size_t ws_size, hipStream_t stream) {
  (void)in_sizes; (void)n_in; (void)out_size; (void)ws_size;
  const float* x    = (const float*)d_in[0];
  const float* ln1g = (const float*)d_in[1];
  const float* ln1b = (const float*)d_in[2];
  const float* ln2g = (const float*)d_in[3];
  const float* ln2b = (const float*)d_in[4];
  const float* Wv   = (const float*)d_in[9];
  const float* bv   = (const float*)d_in[10];
  const float* Wi   = (const float*)d_in[11];
  const float* bi   = (const float*)d_in[12];
  const float* Wf   = (const float*)d_in[13];
  const float* bfb  = (const float*)d_in[14];
  const float* Wo   = (const float*)d_in[15];
  const float* bo   = (const float*)d_in[16];
  const float* W1   = (const float*)d_in[17];
  const float* b1   = (const float*)d_in[18];
  const float* W2   = (const float*)d_in[19];
  const float* b2   = (const float*)d_in[20];
  float* xout = (float*)d_out;  // x_mid in-place, then final output

  char* p = (char*)d_ws;
  auto alloc = [&](size_t bytes) {
    char* r = p;
    p += (bytes + 255) & ~size_t(255);
    return r;
  };
  __bf16* WvT = (__bf16*)alloc((size_t)DD * DD * 2);  // [N][K] transposed
  __bf16* WoT = (__bf16*)alloc((size_t)DD * DD * 2);
  __bf16* W1T = (__bf16*)alloc((size_t)HH * DD * 2);
  __bf16* W2T = (__bf16*)alloc((size_t)DD * HH * 2);
  __bf16* h_bf  = (__bf16*)alloc((size_t)BB * SS * DD * 2);
  __bf16* vT_bf = (__bf16*)alloc((size_t)BB * DD * SS * 2);  // [B][D][S]
  __bf16* h2_bf = (__bf16*)alloc((size_t)BB * SS * DD * 2);
  __bf16* m_bf  = (__bf16*)alloc((size_t)BB * SS * HH * 2);
  float* o_f = (float*)alloc((size_t)BB * SS * DD * 4);
  float* fl  = (float*)alloc((size_t)BB * SS * 4);
  float* il  = (float*)alloc((size_t)BB * SS * 4);
  float* lfc = (float*)alloc((size_t)BB * SS * 4);
  float* wv_ = (float*)alloc((size_t)BB * SS * 4);

  // Weight transpose+convert (amortized: each weight reused 64-128x/call).
  cvt_tr_kernel<<<dim3(DD / 32, DD / 32), 256, 0, stream>>>(Wv, WvT, DD, DD);
  cvt_tr_kernel<<<dim3(DD / 32, DD / 32), 256, 0, stream>>>(Wo, WoT, DD, DD);
  cvt_tr_kernel<<<dim3(HH / 32, DD / 32), 256, 0, stream>>>(W1, W1T, DD, HH);
  cvt_tr_kernel<<<dim3(DD / 32, HH / 32), 256, 0, stream>>>(W2, W2T, HH, DD);

  // LN1 + gate matvecs, then per-batch cumulative gate scan.
  ln_kernel<<<BB * SS, 256, 0, stream>>>(x, ln1g, ln1b, h_bf, Wf, Wi, fl, il, 1);
  gate_scan_kernel<<<BB, 256, 0, stream>>>(fl, il, bfb, bi, lfc, wv_);

  // v = h@Wv+bv, written transposed for attention; o = sigmoid(h@Wo+bo).
  gemm_bf16_kernel<0, 2, false>
      <<<dim3(DD / 64, (BB * SS) / 128), 256, 0, stream>>>(
          h_bf, WvT, bv, nullptr, nullptr, vT_bf, BB * SS, DD, DD);
  gemm_bf16_kernel<1, 0, false>
      <<<dim3(DD / 64, (BB * SS) / 128), 256, 0, stream>>>(
          h_bf, WoT, bo, nullptr, o_f, nullptr, BB * SS, DD, DD);

  // Fused attention + gated residual -> xout (= x_mid).
  attn_kernel<<<dim3(DD / 256, SS / 16, BB), 256, 0, stream>>>(
      vT_bf, lfc, wv_, x, o_f, xout);

  // LN2, MLP up (LeakyReLU^2 fused), MLP down with residual add in-place.
  ln_kernel<<<BB * SS, 256, 0, stream>>>(xout, ln2g, ln2b, h2_bf, nullptr,
                                         nullptr, nullptr, nullptr, 0);
  gemm_bf16_kernel<2, 1, false>
      <<<dim3(HH / 64, (BB * SS) / 128), 256, 0, stream>>>(
          h2_bf, W1T, b1, nullptr, nullptr, m_bf, BB * SS, DD, HH);
  gemm_bf16_kernel<0, 0, true>
      <<<dim3(DD / 64, (BB * SS) / 128), 256, 0, stream>>>(
          m_bf, W2T, b2, xout, xout, nullptr, BB * SS, HH, DD);
}
